// TPSRegistration_5789615915077
// MI455X (gfx1250) — matrix operand order repaired and checked by reading, not compiled, not run on hardware
//
#include <hip/hip_runtime.h>
#include <hip/hip_bf16.h>
#include <math.h>

// CDNA5 wave32 WMMA types
typedef __attribute__((ext_vector_type(16))) _Float16 v16h;
typedef __attribute__((ext_vector_type(8)))  _Float16 v8h_t;
typedef __attribute__((ext_vector_type(8)))  float    v8f;

// Async global->LDS path (CDNA5 GLOBAL_LOAD_ASYNC_TO_LDS_B128, ASYNCcnt).
// Builtin signature (from hipcc diagnostic): (int4 AS1*, int4 AS3*, Imm, Imm)
typedef int v4i_ __attribute__((vector_size(16)));
typedef __attribute__((address_space(1))) v4i_ Gv4i;
typedef __attribute__((address_space(3))) v4i_ Lv4i;

#if defined(__has_builtin)
#if __has_builtin(__builtin_amdgcn_global_load_async_to_lds_b128)
#define ATHENA_ASYNC_LDS 1
#endif
#if __has_builtin(__builtin_amdgcn_s_wait_asynccnt)
#define ATHENA_WAIT_ASYNC() __builtin_amdgcn_s_wait_asynccnt(0)
#else
#define ATHENA_WAIT_ASYNC() asm volatile("s_wait_asynccnt 0x0" ::: "memory")
#endif
#else
#define ATHENA_WAIT_ASYNC() asm volatile("s_wait_asynccnt 0x0" ::: "memory")
#endif

// Problem constants (from reference setup_inputs)
// B=16, C=64 (concat -> 128), H=W=128, hid=128, K=5 (25 ctrl pts, L is 28x28)

// ---------------------------------------------------------------------------
// Kernel 1: pack concat(src,tgt) NCHW f32 -> NHWC f16 (LDS transpose per row)
// ---------------------------------------------------------------------------
__global__ void k_pack_x(const float* __restrict__ src, const float* __restrict__ tgt,
                         _Float16* __restrict__ xf) {
  __shared__ _Float16 tile[128 * 136];   // [x][c], padded stride to dodge bank conflicts
  const int y = blockIdx.x;              // 0..127
  const int b = blockIdx.y;              // 0..15
  const int tid = threadIdx.x;           // 0..255
  // read phase: coalesced over x for each channel
  for (int i = tid; i < 16384; i += 256) {
    int c = i >> 7, x = i & 127;
    float v = (c < 64)
      ? src[(((size_t)b * 64 + c) << 14) + (y << 7) + x]
      : tgt[(((size_t)b * 64 + (c - 64)) << 14) + (y << 7) + x];
    tile[x * 136 + c] = (_Float16)v;
  }
  __syncthreads();
  // write phase: coalesced over c for each pixel
  _Float16* dst = xf + (((size_t)b * 128 + y) << 14);
  for (int i = tid; i < 16384; i += 256) {
    int x = i >> 7, c = i & 127;
    dst[(x << 7) + c] = tile[x * 136 + c];
  }
}

// ---------------------------------------------------------------------------
// Kernel 2: pack conv_w [cout][cin][7][7] f32 -> [tap][cout][cin] f16
//           (B-matrix column-major: K=cin contiguous per cout). Zero pooled.
// ---------------------------------------------------------------------------
__global__ void k_pack_w(const float* __restrict__ w, _Float16* __restrict__ wf,
                         float* __restrict__ pooled) {
  int idx = blockIdx.x * 256 + threadIdx.x;   // < 49*128*128 = 802816
  int cin  = idx & 127;
  int cout = (idx >> 7) & 127;
  int tap  = idx >> 14;                       // ky*7+kx
  wf[idx] = (_Float16)w[((size_t)(cout * 128 + cin)) * 49 + tap];
  if (idx < 16 * 128) pooled[idx] = 0.f;
}

// ---------------------------------------------------------------------------
// A-tile fill: 2 threads per output pixel, 64 channels (128B) each.
// In-bounds rows stream straight from global into LDS via the async path
// (no VGPR staging); zero-padding columns are plain ds_store zeros.
// ---------------------------------------------------------------------------
__device__ __forceinline__ void fill_tile(_Float16* tile, const _Float16* srow,
                                          int kx, int tid) {
  const int p  = tid >> 1;                  // output x (M row)
  const int hc = (tid & 1) << 6;            // channel base 0/64
  const int xs = p + kx - 3;
  _Float16* d = tile + p * 136 + hc;
  if ((unsigned)xs < 128u) {
    const _Float16* s = srow + (((size_t)xs) << 7) + hc;
#if defined(ATHENA_ASYNC_LDS)
    #pragma unroll
    for (int j = 0; j < 8; ++j)
      __builtin_amdgcn_global_load_async_to_lds_b128(
          (Gv4i*)(s + 8 * j), (Lv4i*)(d + 8 * j), 0, 0);
#else
    #pragma unroll
    for (int j = 0; j < 8; ++j)
      *(v8h_t*)(d + 8 * j) = *(const v8h_t*)(s + 8 * j);
#endif
  } else {
    v8h_t z;
    #pragma unroll
    for (int e = 0; e < 8; ++e) z[e] = (_Float16)0.f;
    #pragma unroll
    for (int j = 0; j < 8; ++j) *(v8h_t*)(d + 8 * j) = z;
  }
}

// ---------------------------------------------------------------------------
// Kernel 3: implicit-GEMM conv (7x7, pad 3) via WMMA f32_16x16x32_f16,
//           double-buffered async LDS staging, fused bias+ReLU+mean.
// One workgroup = (b, y): M=128 pixels x N=128 couts; K = taps * 128 cin.
// 8 waves; wave w owns N-tile w (couts 16w..16w+15), loops 8 M-tiles.
// ---------------------------------------------------------------------------
__global__ void __launch_bounds__(256)
k_conv_pool(const _Float16* __restrict__ xf, const _Float16* __restrict__ wf,
            const float* __restrict__ convb, float* __restrict__ pooled) {
  __shared__ _Float16 aTile[2][128 * 136];   // padded row stride (272B) spreads banks
  const int y    = blockIdx.x;
  const int b    = blockIdx.y;
  const int tid  = threadIdx.x;
  const int wv   = tid >> 5;       // wave 0..7 (wave32)
  const int lane = tid & 31;
  const int lhalf = lane >> 4;     // lane half selects K sub-range (contiguous: K 16h..16h+15)
  const int l16   = lane & 15;

  v8f acc[8];
  #pragma unroll
  for (int t = 0; t < 8; ++t)
    #pragma unroll
    for (int r = 0; r < 8; ++r) acc[t][r] = 0.f;

  // valid ky range (ys = y+ky-3 in [0,128)) is uniform across the workgroup
  const int kyLo = (y < 3) ? (3 - y) : 0;
  const int kyHi = (y > 124) ? (131 - y) : 7;
  const int nT   = (kyHi - kyLo) * 7;        // flattened valid taps

  const _Float16* imgBase = xf + (((size_t)b * 128) << 14);

  // prologue: kick off fill of buffer 0 with the first valid tap
  fill_tile(&aTile[0][0], imgBase + (((size_t)(y + kyLo - 3)) << 14), 0, tid);

  int buf = 0;
  for (int i = 0; i < nT; ++i) {
    ATHENA_WAIT_ASYNC();           // own async fills of aTile[buf] complete
    __syncthreads();               // fills visible; old buffer fully consumed
    if (i + 1 < nT) {
      const int ii = i + 1;
      const int kyN = kyLo + ii / 7;
      const int kxN = ii % 7;
      fill_tile(&aTile[buf ^ 1][0],
                imgBase + (((size_t)(y + kyN - 3)) << 14), kxN, tid);
    }
    const int ky  = kyLo + i / 7;
    const int kx  = i % 7;
    const int tap = ky * 7 + kx;
    // prefetch next tap's weights into cache (global_prefetch_b8)
    if (tap < 48) __builtin_prefetch(wf + (size_t)(tap + 1) * 16384 + tid * 64, 0, 1);
    // ---- compute: 4 K-steps of 32, 8 M-tiles each ----
    const _Float16* wtap = wf + (size_t)tap * 16384 + ((wv * 16 + l16) << 7);
    const _Float16* at   = &aTile[buf][0];
    #pragma unroll
    for (int s = 0; s < 4; ++s) {
      // B frag 32x16: lane half 0 holds K s*32..+15, half 1 holds s*32+16..+31
      v16h bfrag = *(const v16h*)(wtap + s * 32 + (lhalf << 4));
      #pragma unroll
      for (int t = 0; t < 8; ++t) {
        // A frag 16x32: lane l: row 16t + l%16; lane half h holds K 16h..16h+15
        const _Float16* ap = at + (t * 16 + l16) * 136 + s * 32 + (lhalf << 4);
        v8h_t lo = *(const v8h_t*)ap;          // ds_load_b128
        v8h_t hi = *(const v8h_t*)(ap + 8);    // ds_load_b128
        v16h afrag = __builtin_shufflevector(lo, hi,
            0,1,2,3,4,5,6,7,8,9,10,11,12,13,14,15);
        acc[t] = __builtin_amdgcn_wmma_f32_16x16x32_f16(
            false, afrag, false, bfrag, (short)0, acc[t], false, false);
      }
    }
    buf ^= 1;
  }
  // ---- epilogue: bias + ReLU per element, reduce over the 128 pixels ----
  const int n = wv * 16 + l16;                    // output channel
  const float bias = convb[n];
  float ssum = 0.f;
  #pragma unroll
  for (int t = 0; t < 8; ++t)
    #pragma unroll
    for (int r = 0; r < 8; ++r)
      ssum += fmaxf(acc[t][r] + bias, 0.f);
  // lanes l and l+16 hold complementary M halves of the same channel
  ssum += __shfl_xor(ssum, 16, 32);
  if (lane < 16) atomicAdd(&pooled[b * 128 + n], ssum);
}

// ---------------------------------------------------------------------------
// Kernel 4: FC head: h1 = relu(pooled/16384 @ fc1^T + b1); delta = h1@fc2^T+b2
//           ctrl = regular 5x5 grid + delta. Single block.
// ---------------------------------------------------------------------------
__global__ void k_fc_head(const float* __restrict__ pooled,
                          const float* __restrict__ fc1w, const float* __restrict__ fc1b,
                          const float* __restrict__ fc2w, const float* __restrict__ fc2b,
                          float* __restrict__ ctrl_ws, float* __restrict__ ctrl_out) {
  __shared__ float h0[2048];
  __shared__ float h1[2048];
  const int tid = threadIdx.x;
  for (int i = tid; i < 2048; i += 256) h0[i] = pooled[i] * (1.f / 16384.f);
  __syncthreads();
  for (int i = tid; i < 2048; i += 256) {
    int b = i >> 7, o = i & 127;
    float s = fc1b[o];
    const float* wr = fc1w + o * 128;
    const float* hv = h0 + b * 128;
    #pragma unroll 8
    for (int k = 0; k < 128; ++k) s += hv[k] * wr[k];
    h1[i] = fmaxf(s, 0.f);
  }
  __syncthreads();
  for (int i = tid; i < 800; i += 256) {
    int b = i / 50, o = i % 50;
    float s = fc2b[o];
    const float* wr = fc2w + o * 128;
    const float* hv = h1 + b * 128;
    #pragma unroll 8
    for (int k = 0; k < 128; ++k) s += hv[k] * wr[k];
    int kp = o >> 1, d = o & 1;
    float base = (d == 0) ? (-1.f + 0.5f * (float)(kp % 5))
                          : (-1.f + 0.5f * (float)(kp / 5));
    float v = base + s;
    ctrl_ws[i]  = v;     // for TPS kernels
    ctrl_out[i] = v;     // second output of the tuple
  }
}

// ---------------------------------------------------------------------------
// Kernel 5: per-batch 28x28 TPS solve (Gauss-Jordan, partial pivoting)
// ---------------------------------------------------------------------------
__global__ void k_tps_solve(const float* __restrict__ ctrl, float* __restrict__ coef) {
  const int b = threadIdx.x;
  if (b >= 16) return;
  float cx[25], cy[25];
  for (int k = 0; k < 25; ++k) { cx[k] = ctrl[b*50 + 2*k]; cy[k] = ctrl[b*50 + 2*k + 1]; }
  float L[28][28];
  float Y0[28], Y1[28];
  for (int i = 0; i < 25; ++i) {
    for (int j = 0; j < 25; ++j) {
      float dx = cx[i] - cx[j], dy = cy[i] - cy[j];
      float r2 = dx*dx + dy*dy;
      L[i][j] = r2 * __logf(r2 + 1e-6f);
    }
    L[i][25] = 1.f; L[i][26] = cx[i]; L[i][27] = cy[i];
    Y0[i] = cx[i];  Y1[i] = cy[i];
  }
  for (int j = 0; j < 28; ++j) {
    L[25][j] = (j < 25) ? 1.f   : 0.f;
    L[26][j] = (j < 25) ? cx[j] : 0.f;
    L[27][j] = (j < 25) ? cy[j] : 0.f;
  }
  for (int i = 25; i < 28; ++i) { Y0[i] = 0.f; Y1[i] = 0.f; }
  for (int col = 0; col < 28; ++col) {
    int piv = col; float best = fabsf(L[col][col]);
    for (int r = col + 1; r < 28; ++r) {
      float a = fabsf(L[r][col]);
      if (a > best) { best = a; piv = r; }
    }
    if (piv != col) {
      for (int j = 0; j < 28; ++j) { float t = L[col][j]; L[col][j] = L[piv][j]; L[piv][j] = t; }
      float t0 = Y0[col]; Y0[col] = Y0[piv]; Y0[piv] = t0;
      float t1 = Y1[col]; Y1[col] = Y1[piv]; Y1[piv] = t1;
    }
    float inv = 1.f / L[col][col];
    for (int r = 0; r < 28; ++r) {
      if (r == col) continue;
      float f = L[r][col] * inv;
      for (int j = col; j < 28; ++j) L[r][j] -= f * L[col][j];
      Y0[r] -= f * Y0[col];
      Y1[r] -= f * Y1[col];
    }
  }
  for (int i = 0; i < 28; ++i) {
    float inv = 1.f / L[i][i];
    coef[b*56 + 2*i]     = Y0[i] * inv;
    coef[b*56 + 2*i + 1] = Y1[i] * inv;
  }
}

// ---------------------------------------------------------------------------
// Kernel 6: fused TPS grid eval + bilinear zero-padded grid_sample (64 ch)
// ---------------------------------------------------------------------------
__global__ void k_tps_warp(const float* __restrict__ src,
                           const float* __restrict__ ctrl, const float* __restrict__ coef,
                           float* __restrict__ out) {
  __shared__ float sc[50];   // ctrl pts for this batch
  __shared__ float sf[56];   // coeffs [28][2]: rows 0..24 w, 25 a0, 26 a_x, 27 a_y
  const int bIdx = blockIdx.x;           // 1024 blocks, 64 per batch
  const int b = bIdx >> 6;
  const int tid = threadIdx.x;
  if (tid < 50) sc[tid] = ctrl[b*50 + tid];
  else if (tid < 106) sf[tid - 50] = coef[b*56 + (tid - 50)];
  __syncthreads();
  const int p = ((bIdx & 63) << 8) + tid;   // pixel 0..16383
  const int y = p >> 7, x = p & 127;
  const float gx = -1.f + (float)x * (2.f / 127.f);
  const float gy = -1.f + (float)y * (2.f / 127.f);
  float sx = sf[50] + sf[52]*gx + sf[54]*gy;
  float sy = sf[51] + sf[53]*gx + sf[55]*gy;
  #pragma unroll
  for (int k = 0; k < 25; ++k) {
    float dx = sc[2*k] - gx, dy = sc[2*k+1] - gy;
    float r2 = dx*dx + dy*dy;
    float u = r2 * __logf(r2 + 1e-6f);
    sx += sf[2*k] * u;
    sy += sf[2*k+1] * u;
  }
  const float xi = (sx + 1.f) * 64.f - 0.5f;
  const float yi = (sy + 1.f) * 64.f - 0.5f;
  const float x0f = floorf(xi), y0f = floorf(yi);
  const int x0 = (int)x0f, y0 = (int)y0f;
  const int x1 = x0 + 1,  y1 = y0 + 1;
  const float wx1 = xi - x0f, wx0 = 1.f - wx1;
  const float wy1 = yi - y0f, wy0 = 1.f - wy1;
  const float m00 = wy0*wx0 * ((x0 >= 0 && x0 < 128 && y0 >= 0 && y0 < 128) ? 1.f : 0.f);
  const float m01 = wy0*wx1 * ((x1 >= 0 && x1 < 128 && y0 >= 0 && y0 < 128) ? 1.f : 0.f);
  const float m10 = wy1*wx0 * ((x0 >= 0 && x0 < 128 && y1 >= 0 && y1 < 128) ? 1.f : 0.f);
  const float m11 = wy1*wx1 * ((x1 >= 0 && x1 < 128 && y1 >= 0 && y1 < 128) ? 1.f : 0.f);
  const int cx0 = min(max(x0, 0), 127), cx1 = min(max(x1, 0), 127);
  const int cy0 = min(max(y0, 0), 127), cy1 = min(max(y1, 0), 127);
  const float* base = src + ((size_t)b << 20);      // b*64*16384
  const int o00 = (cy0 << 7) + cx0, o01 = (cy0 << 7) + cx1;
  const int o10 = (cy1 << 7) + cx0, o11 = (cy1 << 7) + cx1;
  float* obase = out + ((size_t)b << 20) + (y << 7) + x;
  #pragma unroll 4
  for (int c = 0; c < 64; ++c) {
    const float* ch = base + (c << 14);
    obase[c << 14] = m00*ch[o00] + m01*ch[o01] + m10*ch[o10] + m11*ch[o11];
  }
}

// ---------------------------------------------------------------------------
// Host launcher
// ---------------------------------------------------------------------------
extern "C" void kernel_launch(void* const* d_in, const int* in_sizes, int n_in,
                              void* d_out, int out_size, void* d_ws, size_t ws_size,
                              hipStream_t stream) {
  (void)in_sizes; (void)n_in; (void)out_size; (void)ws_size;
  const float* src   = (const float*)d_in[0];
  const float* tgt   = (const float*)d_in[1];
  const float* convw = (const float*)d_in[2];
  const float* convb = (const float*)d_in[3];
  const float* fc1w  = (const float*)d_in[4];
  const float* fc1b  = (const float*)d_in[5];
  const float* fc2w  = (const float*)d_in[6];
  const float* fc2b  = (const float*)d_in[7];

  char* ws = (char*)d_ws;
  _Float16* xf     = (_Float16*)(ws);               // 16*128*128*128 f16 = 64 MiB
  _Float16* wf     = (_Float16*)(ws + 67108864);    // 49*128*128 f16
  float*    pooled = (float*)(ws + 68714496);       // 16*128 f32
  float*    ctrlw  = (float*)(ws + 68722688);       // 16*25*2 f32
  float*    coef   = (float*)(ws + 68725888);       // 16*28*2 f32

  float* warped = (float*)d_out;                    // 16*64*128*128
  float* ctrlo  = (float*)d_out + 16777216;         // 16*25*2

  k_pack_x   <<<dim3(128, 16), 256, 0, stream>>>(src, tgt, xf);
  k_pack_w   <<<3136,          256, 0, stream>>>(convw, wf, pooled);
  k_conv_pool<<<dim3(128, 16), 256, 0, stream>>>(xf, wf, convb, pooled);
  k_fc_head  <<<1,             256, 0, stream>>>(pooled, fc1w, fc1b, fc2w, fc2b, ctrlw, ctrlo);
  k_tps_solve<<<1,              32, 0, stream>>>(ctrlw, coef);
  k_tps_warp <<<1024,          256, 0, stream>>>(src, ctrlw, coef, warped);
}
